// MEX_41386304864396
// MI455X (gfx1250) — compile-verified, matrix operand order checked by reading, not run on hardware
//
#include <hip/hip_runtime.h>
#include <math.h>

// ---------------------------------------------------------------------------
// MI455X (gfx1250) implementation.
// Strategy: every matmul goes through v_wmma_f32_16x16x32_bf16 (f32 accum,
// bf16 operands converted on the global->LDS staging path). Compute-bound
// workload (~215 GFLOP vs ~100MB traffic), so WMMA utilization is everything.
// ---------------------------------------------------------------------------

typedef __bf16 bf16_t;
typedef __attribute__((ext_vector_type(16))) __bf16 v16bf;
typedef __attribute__((ext_vector_type(8)))  __bf16 v8bf;
typedef __attribute__((ext_vector_type(8)))  float   v8f;

static constexpr int LDS_STRIDE = 40;   // 32 bf16 + 8 bf16 pad (80B rows -> conflict-free b128 reads)

static __device__ __forceinline__ v16bf load_frag(const bf16_t* base) {
  // base points at row start + (lane/16)*8 elements; K 0..7(+off) then 16..23(+off)
  v8bf lo = *(const v8bf*)(base);
  v8bf hi = *(const v8bf*)(base + 16);
  return __builtin_shufflevector(lo, hi, 0,1,2,3,4,5,6,7,8,9,10,11,12,13,14,15);
}

// ---------------------------------------------------------------------------
// Generic batched GEMM:  C[z] = epilogue( scale * A[z] @ B[z] + bias + resid )
//   A: [M,K] row-major, row stride ldA (K contiguous)
//   B: logical [K,N]; if bTrans==0 memory is [K,N] (row stride ldB),
//      if bTrans==1 memory is [N,K] row-major (row stride ldB)  -> Q@K^T case
//   Batch offsets: z -> outer=z/div, inner=z%div; ptr += outer*O + inner*I
// ---------------------------------------------------------------------------
template<int BM, int BN, int WAVES_M, int WAVES_N>
__global__ __launch_bounds__(WAVES_M*WAVES_N*32)
void gemm_bf16_wmma(const float* __restrict__ A, const float* __restrict__ Bm,
                    const float* __restrict__ bias, const float* __restrict__ resid,
                    float* __restrict__ C,
                    int K, long long ldA, long long ldB, long long ldC,
                    int div,
                    long long aO, long long aI,
                    long long bO, long long bI,
                    long long cO, long long cI,
                    int bTrans, float scale, int doGelu)
{
  constexpr int BK   = 32;
  constexpr int NTH  = WAVES_M*WAVES_N*32;
  constexpr int WM_T = BM/(16*WAVES_M);
  constexpr int WN_T = BN/(16*WAVES_N);

  __shared__ bf16_t Alds[2][BM*LDS_STRIDE];
  __shared__ bf16_t Blds[2][BN*LDS_STRIDE];

  const int tid  = threadIdx.x;
  const int wave = tid >> 5;
  const int lane = tid & 31;
  const int l16  = lane & 15;
  const int hi   = lane >> 4;
  const int waveM = (wave / WAVES_N) * (WM_T*16);
  const int waveN = (wave % WAVES_N) * (WN_T*16);

  const int blockM = blockIdx.y * BM;
  const int blockN = blockIdx.x * BN;

  const int z     = blockIdx.z;
  const int outer = z / div;
  const int inner = z - outer*div;
  A  += (long long)outer*aO + (long long)inner*aI;
  Bm += (long long)outer*bO + (long long)inner*bI;
  const long long coff = (long long)outer*cO + (long long)inner*cI;
  C += coff;
  if (resid) resid += coff;

  v8f acc[WM_T][WN_T] = {};

  const int nk = K / BK;

  auto loadStage = [&](int kt, int buf) {
    const int kbase = kt * BK;
    constexpr int F4R = BK/4;                       // 8 float4 per 32-wide row
    {
      constexpr int ITER = (BM*F4R)/NTH;
      #pragma unroll
      for (int it = 0; it < ITER; ++it) {
        const int i  = tid + it*NTH;
        const int r  = i / F4R;
        const int c4 = (i % F4R)*4;
        const float4 v = *(const float4*)(A + (long long)(blockM + r)*ldA + (kbase + c4));
        bf16_t* d = &Alds[buf][r*LDS_STRIDE + c4];
        d[0]=(bf16_t)v.x; d[1]=(bf16_t)v.y; d[2]=(bf16_t)v.z; d[3]=(bf16_t)v.w;
      }
    }
    if (bTrans) {                                   // memory [N,K]: same pattern as A
      constexpr int ITER = (BN*F4R)/NTH;
      #pragma unroll
      for (int it = 0; it < ITER; ++it) {
        const int i  = tid + it*NTH;
        const int r  = i / F4R;
        const int c4 = (i % F4R)*4;
        const float4 v = *(const float4*)(Bm + (long long)(blockN + r)*ldB + (kbase + c4));
        bf16_t* d = &Blds[buf][r*LDS_STRIDE + c4];
        d[0]=(bf16_t)v.x; d[1]=(bf16_t)v.y; d[2]=(bf16_t)v.z; d[3]=(bf16_t)v.w;
      }
    } else {                                        // memory [K,N]: coalesced read, transposed LDS write
      constexpr int NF4  = BN/4;
      constexpr int ITER = (BK*NF4)/NTH;
      #pragma unroll
      for (int it = 0; it < ITER; ++it) {
        const int i  = tid + it*NTH;
        const int k  = i / NF4;
        const int n4 = (i % NF4)*4;
        const float4 v = *(const float4*)(Bm + (long long)(kbase + k)*ldB + (blockN + n4));
        Blds[buf][(n4+0)*LDS_STRIDE + k] = (bf16_t)v.x;
        Blds[buf][(n4+1)*LDS_STRIDE + k] = (bf16_t)v.y;
        Blds[buf][(n4+2)*LDS_STRIDE + k] = (bf16_t)v.z;
        Blds[buf][(n4+3)*LDS_STRIDE + k] = (bf16_t)v.w;
      }
    }
  };

  loadStage(0, 0);
  __syncthreads();

  for (int kt = 0; kt < nk; ++kt) {
    if (kt + 1 < nk) loadStage(kt + 1, (kt + 1) & 1);
    const int buf = kt & 1;

    v16bf afrag[WM_T];
    v16bf bfrag[WN_T];
    #pragma unroll
    for (int i = 0; i < WM_T; ++i)
      afrag[i] = load_frag(&Alds[buf][(waveM + i*16 + l16)*LDS_STRIDE + hi*8]);
    #pragma unroll
    for (int j = 0; j < WN_T; ++j)
      bfrag[j] = load_frag(&Blds[buf][(waveN + j*16 + l16)*LDS_STRIDE + hi*8]);

    #pragma unroll
    for (int i = 0; i < WM_T; ++i)
      #pragma unroll
      for (int j = 0; j < WN_T; ++j)
        acc[i][j] = __builtin_amdgcn_wmma_f32_16x16x32_bf16(
            false, afrag[i], false, bfrag[j], (short)0, acc[i][j], false, false);

    __syncthreads();
  }

  // Epilogue: C layout per ISA: lane -> n = l16, VGPR r -> m = r + 8*hi
  #pragma unroll
  for (int i = 0; i < WM_T; ++i) {
    #pragma unroll
    for (int j = 0; j < WN_T; ++j) {
      const int gm0 = blockM + waveM + i*16 + hi*8;
      const int gn  = blockN + waveN + j*16 + l16;
      const float bv = bias ? bias[gn] : 0.0f;
      #pragma unroll
      for (int r = 0; r < 8; ++r) {
        const long long off = (long long)(gm0 + r)*ldC + gn;
        float v = acc[i][j][r]*scale + bv;
        if (resid)  v += resid[off];
        if (doGelu) v = 0.5f*v*(1.0f + erff(v*0.70710678118654752f));
        C[off] = v;
      }
    }
  }
}

// ---------------------------------------------------------------------------
// [S,B,D] -> [B,S,D] permute (D=1024, one row per block)
// ---------------------------------------------------------------------------
__global__ __launch_bounds__(256)
void permute_sbd_bsd(const float* __restrict__ in, float* __restrict__ out, int S, int Bn)
{
  const int sb = blockIdx.x;            // s-major
  const int s  = sb / Bn;
  const int b  = sb % Bn;
  const int t4 = threadIdx.x * 4;
  const float4 v = *(const float4*)(in + ((long long)(s*Bn + b))*1024 + t4);
  *(float4*)(out + ((long long)(b*S + s))*1024 + t4) = v;
}

// ---------------------------------------------------------------------------
// Row LayerNorm over D=1024 (one row per 256-thread block, 4 elems/thread)
// ---------------------------------------------------------------------------
__global__ __launch_bounds__(256)
void layernorm_k(const float* __restrict__ x, const float* __restrict__ gamma,
                 const float* __restrict__ beta, float* __restrict__ y)
{
  __shared__ float red[256];
  const int tid = threadIdx.x;
  const long long base = (long long)blockIdx.x*1024 + tid*4;
  const float4 v = *(const float4*)(x + base);

  red[tid] = v.x + v.y + v.z + v.w;
  __syncthreads();
  for (int o = 128; o > 0; o >>= 1) { if (tid < o) red[tid] += red[tid+o]; __syncthreads(); }
  const float mean = red[0] * (1.0f/1024.0f);
  __syncthreads();

  float4 d; d.x=v.x-mean; d.y=v.y-mean; d.z=v.z-mean; d.w=v.w-mean;
  red[tid] = d.x*d.x + d.y*d.y + d.z*d.z + d.w*d.w;
  __syncthreads();
  for (int o = 128; o > 0; o >>= 1) { if (tid < o) red[tid] += red[tid+o]; __syncthreads(); }
  const float rs = rsqrtf(red[0]*(1.0f/1024.0f) + 1e-5f);

  const float4 g = *(const float4*)(gamma + tid*4);
  const float4 b = *(const float4*)(beta  + tid*4);
  float4 o;
  o.x = d.x*rs*g.x + b.x;
  o.y = d.y*rs*g.y + b.y;
  o.z = d.z*rs*g.z + b.z;
  o.w = d.w*rs*g.w + b.w;
  *(float4*)(y + base) = o;
}

// ---------------------------------------------------------------------------
// In-place row softmax, row length 512 (2 elems/thread)
// ---------------------------------------------------------------------------
__global__ __launch_bounds__(256)
void softmax_k(float* __restrict__ p)
{
  __shared__ float red[256];
  const int tid = threadIdx.x;
  float* x = p + (long long)blockIdx.x*512;
  float2 v = *(float2*)(x + tid*2);

  red[tid] = fmaxf(v.x, v.y);
  __syncthreads();
  for (int o = 128; o > 0; o >>= 1) { if (tid < o) red[tid] = fmaxf(red[tid], red[tid+o]); __syncthreads(); }
  const float m = red[0];
  __syncthreads();

  v.x = __expf(v.x - m);
  v.y = __expf(v.y - m);
  red[tid] = v.x + v.y;
  __syncthreads();
  for (int o = 128; o > 0; o >>= 1) { if (tid < o) red[tid] += red[tid+o]; __syncthreads(); }
  const float inv = 1.0f / red[0];
  v.x *= inv; v.y *= inv;
  *(float2*)(x + tid*2) = v;
}

// ---------------------------------------------------------------------------
// Host-side GEMM dispatch
// ---------------------------------------------------------------------------
static void run_gemm(hipStream_t st,
                     const float* A, const float* B, const float* bias, const float* resid, float* C,
                     int M, int N, int K,
                     long long ldA, long long ldB, long long ldC,
                     int batches, int div,
                     long long aO, long long aI, long long bO, long long bI,
                     long long cO, long long cI,
                     int bTrans, float scale, int gelu)
{
  if (N % 128 == 0) {
    dim3 g(N/128, M/128, batches), b(256);
    gemm_bf16_wmma<128,128,2,4><<<g,b,0,st>>>(A,B,bias,resid,C,K,ldA,ldB,ldC,div,
                                              aO,aI,bO,bI,cO,cI,bTrans,scale,gelu);
  } else {  // N == 64 (per-head context GEMMs)
    dim3 g(N/64, M/128, batches), b(256);
    gemm_bf16_wmma<128,64,4,2><<<g,b,0,st>>>(A,B,bias,resid,C,K,ldA,ldB,ldC,div,
                                             aO,aI,bO,bI,cO,cI,bTrans,scale,gelu);
  }
}

extern "C" void kernel_launch(void* const* d_in, const int* in_sizes, int n_in,
                              void* d_out, int out_size, void* d_ws, size_t ws_size,
                              hipStream_t stream)
{
  (void)in_sizes; (void)n_in; (void)out_size; (void)ws_size;

  const float* global_feat = (const float*)d_in[0];
  const float* local_feat  = (const float*)d_in[1];
  const float* text_feat   = (const float*)d_in[2];
  const float* Wg_emb = (const float*)d_in[3];
  const float* Wl1    = (const float*)d_in[4];
  const float* Wl2    = (const float*)d_in[5];
  const float* Wq_g   = (const float*)d_in[6];
  const float* Wk_g   = (const float*)d_in[7];
  const float* Wv_g   = (const float*)d_in[8];
  const float* Wq_p   = (const float*)d_in[9];
  const float* Wk_p   = (const float*)d_in[10];
  const float* Wv_p   = (const float*)d_in[11];
  const float* Wd     = (const float*)d_in[12];
  const float* Wml    = (const float*)d_in[13];
  const float* bg_emb = (const float*)d_in[14];
  const float* bl1    = (const float*)d_in[15];
  const float* bl2    = (const float*)d_in[16];
  const float* bq_g   = (const float*)d_in[17];
  const float* bk_g   = (const float*)d_in[18];
  const float* bv_g   = (const float*)d_in[19];
  const float* bq_p   = (const float*)d_in[20];
  const float* bk_p   = (const float*)d_in[21];
  const float* bv_p   = (const float*)d_in[22];
  const float* bd     = (const float*)d_in[23];
  const float* bml    = (const float*)d_in[24];
  const float* bproj  = (const float*)d_in[25];
  const float* betag  = (const float*)d_in[26];
  const float* betal1 = (const float*)d_in[27];
  const float* betal2 = (const float*)d_in[28];
  const float* gg     = (const float*)d_in[29];
  const float* gl1    = (const float*)d_in[30];
  const float* gl2    = (const float*)d_in[31];
  const float* Wfc    = (const float*)d_in[32];
  const float* bfc    = (const float*)d_in[33];
  const float* Wproj  = (const float*)d_in[34];

  const int S = 512, Bn = 8, D = 1024, H = 16, DH = 64;
  const int rows = Bn * S;                 // 4096
  const long long SD  = (long long)S*D;    // per-batch activation stride
  const long long F   = (long long)rows*D; // 4 Mi floats (16 MiB)
  const long long SS  = (long long)S*S;
  const long long HSS = (long long)H*SS;

  float* ws    = (float*)d_ws;             // needs ~39*F floats (~624 MiB)
  float* o_g    = ws + 0*F;
  float* o_l    = ws + 1*F;
  float* o_t    = ws + 2*F;
  float* o_tmp  = ws + 3*F;
  float* o_dup  = ws + 4*F;
  float* o_l2   = ws + 5*F;
  float* o_g2   = ws + 6*F;
  float* o_qg   = ws + 7*F;
  float* o_kg   = ws + 8*F;
  float* o_vg   = ws + 9*F;
  float* o_qp   = ws + 10*F;
  float* o_kp   = ws + 11*F;
  float* o_vp   = ws + 12*F;
  float* o_glb  = ws + 13*F;               // B*H*S*S = 8F
  float* o_plb  = ws + 21*F;               // 8F
  float* o_enh  = ws + 29*F;               // 8F
  float* o_ctxp = ws + 37*F;
  float* o_sum  = ws + 38*F;
  float* o_d1   = o_tmp;                   // reuse (free after g2 LN)
  float* o_fc   = ws + 13*F;               // reuse glb+plb region (16F) after attention
  float* o_proj = o_dup;                   // reuse (free after qp projection)
  float* outp   = (float*)d_out;

  // ---- permute [S,B,D] -> [B,S,D] --------------------------------------
  permute_sbd_bsd<<<dim3(S*Bn), dim3(256), 0, stream>>>(global_feat, o_g, S, Bn);
  permute_sbd_bsd<<<dim3(S*Bn), dim3(256), 0, stream>>>(local_feat,  o_l, S, Bn);
  permute_sbd_bsd<<<dim3(S*Bn), dim3(256), 0, stream>>>(text_feat,   o_t, S, Bn);

  // ---- FFN embeddings: LN(x + x@W + b) ---------------------------------
  run_gemm(stream, o_l, Wl1, bl1, o_l, o_tmp, rows, D, D, D, D, D,
           1,1, 0,0, 0,0, 0,0, 0, 1.0f, 0);
  layernorm_k<<<dim3(rows), dim3(256), 0, stream>>>(o_tmp, gl1, betal1, o_dup);

  run_gemm(stream, o_l, Wl2, bl2, o_l, o_tmp, rows, D, D, D, D, D,
           1,1, 0,0, 0,0, 0,0, 0, 1.0f, 0);
  layernorm_k<<<dim3(rows), dim3(256), 0, stream>>>(o_tmp, gl2, betal2, o_l2);

  run_gemm(stream, o_g, Wg_emb, bg_emb, o_g, o_tmp, rows, D, D, D, D, D,
           1,1, 0,0, 0,0, 0,0, 0, 1.0f, 0);
  layernorm_k<<<dim3(rows), dim3(256), 0, stream>>>(o_tmp, gg, betag, o_g2);

  // ---- QKV projections --------------------------------------------------
  run_gemm(stream, o_g2,  Wq_g, bq_g, nullptr, o_qg, rows, D, D, D, D, D, 1,1, 0,0,0,0,0,0, 0, 1.0f, 0);
  run_gemm(stream, o_l2,  Wk_g, bk_g, nullptr, o_kg, rows, D, D, D, D, D, 1,1, 0,0,0,0,0,0, 0, 1.0f, 0);
  run_gemm(stream, o_l2,  Wv_g, bv_g, nullptr, o_vg, rows, D, D, D, D, D, 1,1, 0,0,0,0,0,0, 0, 1.0f, 0);
  run_gemm(stream, o_dup, Wq_p, bq_p, nullptr, o_qp, rows, D, D, D, D, D, 1,1, 0,0,0,0,0,0, 0, 1.0f, 0);
  run_gemm(stream, o_t,   Wk_p, bk_p, nullptr, o_kp, rows, D, D, D, D, D, 1,1, 0,0,0,0,0,0, 0, 1.0f, 0);
  run_gemm(stream, o_t,   Wv_p, bv_p, nullptr, o_vp, rows, D, D, D, D, D, 1,1, 0,0,0,0,0,0, 0, 1.0f, 0);

  // ---- attention scores (batched over B*H, head-strided Q/K, B^T mode) --
  const float scale = 0.125f;  // 1/sqrt(64)
  run_gemm(stream, o_qg, o_kg, nullptr, nullptr, o_glb, S, S, DH, D, D, S,
           Bn*H, H, SD, DH, SD, DH, HSS, SS, 1, scale, 0);
  softmax_k<<<dim3(Bn*H*S), dim3(256), 0, stream>>>(o_glb);

  run_gemm(stream, o_qp, o_kp, nullptr, nullptr, o_plb, S, S, DH, D, D, S,
           Bn*H, H, SD, DH, SD, DH, HSS, SS, 1, scale, 0);
  softmax_k<<<dim3(Bn*H*S), dim3(256), 0, stream>>>(o_plb);

  // ---- enhanced = glb @ plb  (128 batches of 512x512x512) ---------------
  run_gemm(stream, o_glb, o_plb, nullptr, nullptr, o_enh, S, S, S, S, S, S,
           Bn*H, H, HSS, SS, HSS, SS, HSS, SS, 0, 1.0f, 0);

  // ---- contexts: ctx_p = enh @ vp ; sum = glb @ vg + ctx_p ---------------
  run_gemm(stream, o_enh, o_vp, nullptr, nullptr, o_ctxp, S, DH, S, S, D, D,
           Bn*H, H, HSS, SS, SD, DH, SD, DH, 0, 1.0f, 0);
  run_gemm(stream, o_glb, o_vg, nullptr, o_ctxp, o_sum, S, DH, S, S, D, D,
           Bn*H, H, HSS, SS, SD, DH, SD, DH, 0, 1.0f, 0);

  // ---- output MLP -------------------------------------------------------
  run_gemm(stream, o_sum, Wd,    bd,    nullptr, o_d1,   rows, D,   D,   D,   D,   D,
           1,1, 0,0,0,0,0,0, 0, 1.0f, 0);
  run_gemm(stream, o_d1,  Wfc,   bfc,   nullptr, o_fc,   rows, 4*D, D,   D,   4*D, 4*D,
           1,1, 0,0,0,0,0,0, 0, 1.0f, 1);             // fused exact GELU
  run_gemm(stream, o_fc,  Wproj, bproj, nullptr, o_proj, rows, D,   4*D, 4*D, D,   D,
           1,1, 0,0,0,0,0,0, 0, 1.0f, 0);
  run_gemm(stream, o_proj, Wml,  bml,   nullptr, outp,   rows, D,   D,   D,   D,   D,
           1,1, 0,0,0,0,0,0, 0, 1.0f, 0);
}
